// E_GCL_X_11751030522787
// MI455X (gfx1250) — matrix operand order repaired and checked by reading, not compiled, not run on hardware
//
#include <hip/hip_runtime.h>
#include <hip/hip_bf16.h>

typedef __bf16 bf16;
typedef __attribute__((ext_vector_type(16))) __bf16 v16bf;
typedef __attribute__((ext_vector_type(8)))  float  v8f;
typedef __attribute__((ext_vector_type(8)))  unsigned v8u;

#define B_   2
#define N_   5000
#define E_   160000
#define FIN  128
#define HID  128
#define INE  288      // 2*128 + 16 + 16

#define XS   296      // LDS stride for X tile (>=288, keeps 4B row alignment)
#define HS   136      // LDS stride for hidden tile
#define ZS   264      // LDS stride for node z tile
#define WAVES 2       // waves per block; each wave owns a 32-row M tile

// ---------------- fragment loaders ----------------
// A fragment (16x32 bf16, row-major in LDS with stride S):
// lane L: row = L&15, half = L>>4; pair p holds K = kbase + half*8 + (p&3)*2 + (p&4 ? 16 : 0)
__device__ __forceinline__ v16bf load_a_lds(const bf16* base, int S, int kbase, int lane) {
  int row = lane & 15, half = lane >> 4;
  const bf16* p = base + row * S + kbase + half * 8;
  v8u t;
#pragma unroll
  for (int pr = 0; pr < 8; ++pr) {
    int k = ((pr & 4) ? 16 : 0) + (pr & 3) * 2;
    t[pr] = *reinterpret_cast<const unsigned*>(p + k);
  }
  return __builtin_bit_cast(v16bf, t);
}

// B fragment: packed fragment-linear, 16 bf16 (32B) contiguous per lane.
__device__ __forceinline__ v16bf load_b_pk(const bf16* q) {
  const unsigned* u = reinterpret_cast<const unsigned*>(q);
  v8u t;
#pragma unroll
  for (int i = 0; i < 8; ++i) t[i] = u[i];
  return __builtin_bit_cast(v16bf, t);
}

// Two stacked 16x16 output tiles (M rows 0-15 and 16-31) sharing each B fragment:
// doubles FLOPs per byte of weight traffic from L2.
__device__ __forceinline__ void wmma_tile2(const bf16* A, int S, int KT,
                                           const bf16* Wp, int NT, int nt,
                                           float bb, int lane,
                                           v8f& acc0, v8f& acc1) {
#pragma unroll
  for (int r = 0; r < 8; ++r) { acc0[r] = bb; acc1[r] = bb; }
  for (int kt = 0; kt < KT; ++kt) {
    v16bf b  = load_b_pk(Wp + (((long)(kt * NT + nt) * 32 + lane) * 16));
    v16bf a0 = load_a_lds(A,          S, kt * 32, lane);
    v16bf a1 = load_a_lds(A + 16 * S, S, kt * 32, lane);
    acc0 = __builtin_amdgcn_wmma_f32_16x16x32_bf16(false, a0, false, b, (short)0, acc0,
                                                   false, false);
    acc1 = __builtin_amdgcn_wmma_f32_16x16x32_bf16(false, a1, false, b, (short)0, acc1,
                                                   false, false);
  }
}

// ---------------- weight packing (f32 row-major [K,Nsrc] -> bf16 fragment-linear) ----------------
__global__ void pack_w(const float* __restrict__ W, bf16* __restrict__ out,
                       int K, int Nsrc, int Ndst) {
  int i = blockIdx.x * blockDim.x + threadIdx.x;
  int total = K * Ndst;
  if (i >= total) return;
  int e    = i & 15;
  int lane = (i >> 4) & 31;
  int f    = i >> 9;
  int NT   = Ndst >> 4;
  int kt   = f / NT, nt = f % NT;
  int kk = kt * 32 + (lane >> 4) * 8 + (e & 7) + (e >> 3) * 16;   // B layout: lanes 0-15 K 0..7,16..23; lanes 16-31 K 8..15,24..31
  int nn = nt * 16 + (lane & 15);
  float v = (nn < Nsrc) ? W[(long)kk * Nsrc + nn] : 0.0f;
  out[i] = (bf16)v;
}

// ---------------- edge kernel: radial + edge MLP + coord head, fused; 32 edges/wave ----------------
__global__ __launch_bounds__(64) void egcl_edge(
    const float* __restrict__ h, const float* __restrict__ coord,
    const int* __restrict__ ei, const float* __restrict__ ea,
    const bf16* __restrict__ We1p, const float* __restrict__ be1,
    const bf16* __restrict__ We2p, const float* __restrict__ be2,
    const bf16* __restrict__ Wc1p, const float* __restrict__ bc1,
    const bf16* __restrict__ Wc2p,
    float* __restrict__ agg_h, float* __restrict__ agg_c, float* __restrict__ cnt)
{
  __shared__ bf16 sX[WAVES][32][XS];
  __shared__ bf16 sH[WAVES][32][HS];
  __shared__ float sCD[WAVES][32][12];
  __shared__ int  sRow[WAVES][32];
  __shared__ int  sCol[WAVES][32];

  int w = threadIdx.x >> 5, lane = threadIdx.x & 31;
  int t = blockIdx.x * WAVES + w;          // 0 .. B*E/32-1 (exact)
  int batch = t / (E_ / 32);
  int e0 = (t % (E_ / 32)) * 32;

  const float* h_b   = h     + (long)batch * N_ * FIN;
  const float* c_b   = coord + (long)batch * N_ * 12;
  float* aggh_b = agg_h + (long)batch * N_ * HID;
  float* aggc_b = agg_c + (long)batch * N_ * 12;
  float* cnt_b  = cnt   + (long)batch * N_;

  // --- per-edge geometry: cd, Gram, normalized radial; edge_attr; degree count (1 edge/lane) ---
  {
    int e = e0 + lane;
    int rI = ei[e], cI = ei[E_ + e];
    sRow[w][lane] = rI; sCol[w][lane] = cI;
    float cd[12];
#pragma unroll
    for (int j = 0; j < 12; ++j) {
      cd[j] = c_b[(long)rI * 12 + j] - c_b[(long)cI * 12 + j];
      sCD[w][lane][j] = cd[j];
    }
    float prod[16]; float ss = 0.f;
#pragma unroll
    for (int j = 0; j < 4; ++j)
#pragma unroll
      for (int k = 0; k < 4; ++k) {
        float p = cd[j*3]*cd[k*3] + cd[j*3+1]*cd[k*3+1] + cd[j*3+2]*cd[k*3+2];
        prod[j*4+k] = p; ss += p * p;
      }
    float inv = 1.f / fmaxf(sqrtf(ss), 1e-12f);
#pragma unroll
    for (int j = 0; j < 16; ++j) sX[w][lane][256+j] = (bf16)(prod[j] * inv);
#pragma unroll
    for (int j = 0; j < 16; ++j) sX[w][lane][272+j] = (bf16)ea[(long)e * 16 + j];
    atomicAdd(&cnt_b[rI], 1.0f);
  }
  __syncthreads();

  // --- gather h[row], h[col] into X tile (bf16) ---
  for (int i = lane; i < 32 * 128; i += 32) {
    int r = i >> 7, c = i & 127;
    sX[w][r][c]       = (bf16)h_b[(long)sRow[w][r] * FIN + c];
    sX[w][r][128 + c] = (bf16)h_b[(long)sCol[w][r] * FIN + c];
  }
  __syncthreads();

  const bf16* X0 = &sX[w][0][0];
  bf16* H0 = &sH[w][0][0];
  int nlo = lane & 15, half = lane >> 4;
  v8f a0, a1;

  // --- edge layer 1: [32,288]x[288,128], ReLU -> sH ---
#pragma unroll
  for (int nt = 0; nt < 8; ++nt) {
    wmma_tile2(X0, XS, 9, We1p, 8, nt, be1[nt*16 + nlo], lane, a0, a1);
#pragma unroll
    for (int r = 0; r < 8; ++r) {
      H0[(r + 8*half) * HS + nt*16 + nlo]        = (bf16)fmaxf(a0[r], 0.f);
      H0[(16 + r + 8*half) * HS + nt*16 + nlo]   = (bf16)fmaxf(a1[r], 0.f);
    }
  }
  __syncthreads();

  // --- edge layer 2: ReLU -> ef; atomically aggregate, keep bf16 copy in sX ---
#pragma unroll
  for (int nt = 0; nt < 8; ++nt) {
    wmma_tile2(H0, HS, 4, We2p, 8, nt, be2[nt*16 + nlo], lane, a0, a1);
#pragma unroll
    for (int r = 0; r < 8; ++r) {
      int M0 = r + 8*half, M1 = 16 + M0;
      float v0 = fmaxf(a0[r], 0.f), v1 = fmaxf(a1[r], 0.f);
      sX[w][M0][nt*16 + nlo] = (bf16)v0;
      sX[w][M1][nt*16 + nlo] = (bf16)v1;
      atomicAdd(&aggh_b[(long)sRow[w][M0] * HID + nt*16 + nlo], v0);
      atomicAdd(&aggh_b[(long)sRow[w][M1] * HID + nt*16 + nlo], v1);
    }
  }
  __syncthreads();

  // --- coord MLP layer 1: relu(ef @ Wc1 + bc1) -> sH ---
#pragma unroll
  for (int nt = 0; nt < 8; ++nt) {
    wmma_tile2(X0, XS, 4, Wc1p, 8, nt, bc1[nt*16 + nlo], lane, a0, a1);
#pragma unroll
    for (int r = 0; r < 8; ++r) {
      H0[(r + 8*half) * HS + nt*16 + nlo]      = (bf16)fmaxf(a0[r], 0.f);
      H0[(16 + r + 8*half) * HS + nt*16 + nlo] = (bf16)fmaxf(a1[r], 0.f);
    }
  }
  __syncthreads();

  // --- coord head: w = C1 @ Wc2 (N padded 4->16); trans = cd * w; aggregate ---
  {
    wmma_tile2(H0, HS, 4, Wc2p, 1, 0, 0.f, lane, a0, a1);
    if (nlo < 4) {
#pragma unroll
      for (int r = 0; r < 8; ++r) {
        int M0 = r + 8*half, M1 = 16 + M0;
        int n0 = sRow[w][M0], n1 = sRow[w][M1];
#pragma unroll
        for (int tt = 0; tt < 3; ++tt) {
          atomicAdd(&aggc_b[(long)n0 * 12 + nlo*3 + tt], sCD[w][M0][nlo*3 + tt] * a0[r]);
          atomicAdd(&aggc_b[(long)n1 * 12 + nlo*3 + tt], sCD[w][M1][nlo*3 + tt] * a1[r]);
        }
      }
    }
  }
}

// ---------------- node kernel: h_new = h + relu([h,agg_h]@Wn1+b)@Wn2 + b; 32 nodes/wave ----------------
__global__ __launch_bounds__(64) void egcl_node(
    const float* __restrict__ h, const float* __restrict__ agg_h,
    const bf16* __restrict__ Wn1p, const float* __restrict__ bn1,
    const bf16* __restrict__ Wn2p, const float* __restrict__ bn2,
    float* __restrict__ out_h)
{
  __shared__ bf16 sZ[WAVES][32][ZS];
  __shared__ bf16 sHn[WAVES][32][HS];
  const int TPB = (N_ + 31) / 32;   // 157
  const int TOT = TPB * B_;         // 314

  int w = threadIdx.x >> 5, lane = threadIdx.x & 31;
  int t = blockIdx.x * WAVES + w;
  bool valid = t < TOT;
  int tc = valid ? t : 0;
  int batch = tc / TPB;
  int n0 = (tc % TPB) * 32;

  const float* h_b = h     + (long)batch * N_ * FIN;
  const float* a_b = agg_h + (long)batch * N_ * HID;

  for (int i = lane; i < 32 * 256; i += 32) {
    int r = i >> 8, c = i & 255;
    int node = n0 + r;
    float v = 0.f;
    if (valid && node < N_)
      v = (c < 128) ? h_b[(long)node * 128 + c] : a_b[(long)node * 128 + (c - 128)];
    sZ[w][r][c] = (bf16)v;
  }
  __syncthreads();

  const bf16* Z0 = &sZ[w][0][0];
  bf16* H0 = &sHn[w][0][0];
  int nlo = lane & 15, half = lane >> 4;
  v8f a0, a1;

#pragma unroll
  for (int nt = 0; nt < 8; ++nt) {
    wmma_tile2(Z0, ZS, 8, Wn1p, 8, nt, bn1[nt*16 + nlo], lane, a0, a1);
#pragma unroll
    for (int r = 0; r < 8; ++r) {
      H0[(r + 8*half) * HS + nt*16 + nlo]      = (bf16)fmaxf(a0[r], 0.f);
      H0[(16 + r + 8*half) * HS + nt*16 + nlo] = (bf16)fmaxf(a1[r], 0.f);
    }
  }
  __syncthreads();

#pragma unroll
  for (int nt = 0; nt < 8; ++nt) {
    wmma_tile2(H0, HS, 4, Wn2p, 8, nt, bn2[nt*16 + nlo], lane, a0, a1);
#pragma unroll
    for (int r = 0; r < 8; ++r) {
      int c = nt*16 + nlo;
      int node0 = n0 + r + 8*half;
      int node1 = node0 + 16;
      if (valid && node0 < N_)
        out_h[(long)batch * N_ * 128 + (long)node0 * 128 + c] =
            h_b[(long)node0 * 128 + c] + a0[r];
      if (valid && node1 < N_)
        out_h[(long)batch * N_ * 128 + (long)node1 * 128 + c] =
            h_b[(long)node1 * 128 + c] + a1[r];
    }
  }
}

// ---------------- coord update ----------------
__global__ void egcl_coord(const float* __restrict__ coord, const float* __restrict__ agg_c,
                           const float* __restrict__ cnt, float* __restrict__ out_c) {
  int i = blockIdx.x * blockDim.x + threadIdx.x;
  if (i >= B_ * N_) return;
  float inv = 1.f / fmaxf(cnt[i], 1.0f);
#pragma unroll
  for (int t = 0; t < 12; ++t)
    out_c[(long)i * 12 + t] = coord[(long)i * 12 + t] + agg_c[(long)i * 12 + t] * inv;
}

// ---------------- launch ----------------
extern "C" void kernel_launch(void* const* d_in, const int* in_sizes, int n_in,
                              void* d_out, int out_size, void* d_ws, size_t ws_size,
                              hipStream_t stream) {
  const float* h     = (const float*)d_in[0];
  const float* coord = (const float*)d_in[1];
  const int*   ei    = (const int*)d_in[2];
  const float* ea    = (const float*)d_in[3];
  const float* We1 = (const float*)d_in[4];  const float* be1 = (const float*)d_in[5];
  const float* We2 = (const float*)d_in[6];  const float* be2 = (const float*)d_in[7];
  const float* Wn1 = (const float*)d_in[8];  const float* bn1 = (const float*)d_in[9];
  const float* Wn2 = (const float*)d_in[10]; const float* bn2 = (const float*)d_in[11];
  const float* Wc1 = (const float*)d_in[12]; const float* bc1 = (const float*)d_in[13];
  const float* Wc2 = (const float*)d_in[14];

  char* ws = (char*)d_ws;
  bf16* We1p = (bf16*)(ws + 0);        // 288*128*2 = 73728
  bf16* We2p = (bf16*)(ws + 73728);    // 128*128*2 = 32768
  bf16* Wn1p = (bf16*)(ws + 106496);   // 256*128*2 = 65536
  bf16* Wn2p = (bf16*)(ws + 172032);   // 32768
  bf16* Wc1p = (bf16*)(ws + 204800);   // 32768
  bf16* Wc2p = (bf16*)(ws + 237568);   // 128*16*2 = 4096
  const size_t ACC = 241664;
  float* agg_h = (float*)(ws + ACC);                      // B*N*128*4 = 5,120,000
  float* agg_c = (float*)(ws + ACC + 5120000);            // B*N*12*4  =   480,000
  float* cnt   = (float*)(ws + ACC + 5120000 + 480000);   // B*N*4     =    40,000

  hipMemsetAsync(ws + ACC, 0, 5120000 + 480000 + 40000, stream);

  auto packLaunch = [&](const float* W, bf16* out, int K, int Nsrc, int Ndst) {
    int total = K * Ndst;
    pack_w<<<(total + 255) / 256, 256, 0, stream>>>(W, out, K, Nsrc, Ndst);
  };
  packLaunch(We1, We1p, 288, 128, 128);
  packLaunch(We2, We2p, 128, 128, 128);
  packLaunch(Wn1, Wn1p, 256, 128, 128);
  packLaunch(Wn2, Wn2p, 128, 128, 128);
  packLaunch(Wc1, Wc1p, 128, 128, 128);
  packLaunch(Wc2, Wc2p, 128,   4,  16);

  egcl_edge<<<(B_ * (E_ / 32)) / WAVES, 64, 0, stream>>>(
      h, coord, ei, ea, We1p, be1, We2p, be2, Wc1p, bc1, Wc2p,
      agg_h, agg_c, cnt);

  int TOT = ((N_ + 31) / 32) * B_;
  egcl_node<<<(TOT + WAVES - 1) / WAVES, 64, 0, stream>>>(
      h, agg_h, Wn1p, bn1, Wn2p, bn2, (float*)d_out);

  float* out_c = (float*)d_out + (long)B_ * N_ * 128;
  egcl_coord<<<(B_ * N_ + 255) / 256, 256, 0, stream>>>(coord, agg_c, cnt, out_c);
}
